// GAT_17377437680280
// MI455X (gfx1250) — compile-verified, matrix operand order checked by reading, not run on hardware
//
#include <hip/hip_runtime.h>
#include <hip/hip_bf16.h>

// ---------------- problem constants ----------------
#define GAT_IN   100
#define GAT_H    4
#define GAT_D    32
#define GAT_C    47
#define GAT_F    (GAT_H * GAT_D)   // 128
#define GAT_F2   (GAT_H * GAT_C)   // 188
#define GAT_F2P  192               // 188 padded to 16
#define GAT_KP0  128               // 100 padded to 32-multiple
#define GAT_EPS  1e-5f
#define GAT_SLOPE 0.2f

typedef __attribute__((ext_vector_type(16))) __bf16 v16bf;
typedef __attribute__((ext_vector_type(8)))  float  v8f;

union Frag16 { unsigned int u[8]; v16bf v; };

// ---------------- CSR rowptr from sorted dst ----------------
__global__ void build_rowptr(const int* __restrict__ dst, int* __restrict__ rowptr,
                             int E, int N) {
    int e = blockIdx.x * blockDim.x + threadIdx.x;
    if (e >= E) return;
    int d = dst[e];
    int dprev = (e > 0) ? dst[e - 1] : -1;
    if (d != dprev) {
        for (int n = dprev + 1; n <= d; ++n) rowptr[n] = e;
    }
    if (e == E - 1) {
        for (int n = d + 1; n <= N; ++n) rowptr[n] = E;
    }
}

// ---------------- fp32 -> bf16 with K padding (activations) ----------------
__global__ void cvt_pad_bf16(const float* __restrict__ src, __hip_bfloat16* __restrict__ out,
                             int rows, int K, int KP) {
    long idx = (long)blockIdx.x * blockDim.x + threadIdx.x;
    long total = (long)rows * KP;
    if (idx >= total) return;
    int k = (int)(idx % KP);
    long r = idx / KP;
    float v = (k < K) ? src[r * K + k] : 0.f;
    out[idx] = __float2bfloat16(v);
}

// ---------------- weight convert + transpose: W[K,Nc] -> Wt[NcPad,KPad] bf16 ----------------
__global__ void wt_cvt(const float* __restrict__ W, __hip_bfloat16* __restrict__ Wt,
                       int K, int Nc, int KP, int NcP) {
    int idx = blockIdx.x * blockDim.x + threadIdx.x;
    int total = NcP * KP;
    if (idx >= total) return;
    int n = idx / KP;
    int k = idx % KP;
    float v = (n < Nc && k < K) ? W[(long)k * Nc + n] : 0.f;
    Wt[idx] = __float2bfloat16(v);
}

// ---------------- WMMA GEMM, multi-N-tile per wave -----------------------------------------
// C[M,ldc] = A[M,KP](bf16) @ Wt[NcP,KP]^T(bf16) + bias.
// One wave owns a 16-row strip of A and accumulates ALL NT column tiles, so each
// A fragment is loaded once and feeds NT independent v_wmma accumulators.
template <int NT>
__global__ void gemm_wmma_bf16_nt(const __hip_bfloat16* __restrict__ A,
                                  const __hip_bfloat16* __restrict__ Wt,
                                  const float* __restrict__ bias,
                                  float* __restrict__ Cout,
                                  int M, int KP, int Nc, int ldc) {
    int wave = threadIdx.x >> 5;
    int lane = threadIdx.x & 31;
    int tileM = blockIdx.x * (blockDim.x >> 5) + wave;
    int row0 = tileM * 16;
    if (row0 >= M) return;

    int sub = lane & 15;      // A row / B column within tile
    int hi  = lane >> 4;      // K-half selector
    int KPd = KP >> 1;        // row stride in dwords

    int arow = row0 + sub; if (arow >= M) arow = M - 1;
    const unsigned int* Ap = (const unsigned int*)(A + (long)arow * KP);
    const unsigned int* Bbase = (const unsigned int*)Wt + (long)sub * KPd;

    const v8f vzero = {0.f, 0.f, 0.f, 0.f, 0.f, 0.f, 0.f, 0.f};
    v8f acc[NT];
#pragma unroll
    for (int nt = 0; nt < NT; ++nt) acc[nt] = vzero;

    for (int kb = 0; kb < KP; kb += 32) {
        int dw = kb >> 1;                 // dword index of K-chunk base
        Frag16 a;
#pragma unroll
        for (int v = 0; v < 4; ++v) {
            // regs 0..3: K = hi*8 + 2v      -> dword hi*4 + v
            // regs 4..7: K = 16 + hi*8 +2v  -> dword 8 + hi*4 + v
            a.u[v]     = Ap[dw + hi * 4 + v];
            a.u[v + 4] = Ap[dw + 8 + hi * 4 + v];
        }
        __builtin_prefetch(Ap + dw + 16, 0, 3);
#pragma unroll
        for (int nt = 0; nt < NT; ++nt) {
            const unsigned int* Bp = Bbase + (long)nt * 16 * KPd;
            Frag16 b;
#pragma unroll
            for (int v = 0; v < 4; ++v) {
                b.u[v]     = Bp[dw + hi * 4 + v];
                b.u[v + 4] = Bp[dw + 8 + hi * 4 + v];
            }
            acc[nt] = __builtin_amdgcn_wmma_f32_16x16x32_bf16(
                false, a.v, false, b.v, (short)0, acc[nt], false, false);
        }
    }

#pragma unroll
    for (int nt = 0; nt < NT; ++nt) {
        int col = nt * 16 + sub;
        if (col < Nc) {
            float bv = bias ? bias[col] : 0.f;
#pragma unroll
            for (int v = 0; v < 8; ++v) {
                int r = row0 + v + 8 * hi;   // C layout: VGPR v -> row v (+8 for hi half)
                if (r < M) Cout[(long)r * ldc + col] = acc[nt][v] + bv;
            }
        }
    }
}

// ---------------- edge logits: one wave32 per edge, coalesced feature loads ----------------
__global__ void edge_logits_wave(const float* __restrict__ fs, const float* __restrict__ fd,
                                 const int* __restrict__ src, const int* __restrict__ dst,
                                 const float* __restrict__ attn,  // [H, perHead]
                                 float* __restrict__ logits,      // [E, H]
                                 int E, int ld, int perHead) {
    int wpb = blockDim.x >> 5;
    int e = blockIdx.x * wpb + (threadIdx.x >> 5);
    if (e >= E) return;
    int lane = threadIdx.x & 31;
    const float* ps = fs + (long)src[e] * ld;
    const float* pd = fd + (long)dst[e] * ld;
#pragma unroll
    for (int h = 0; h < GAT_H; ++h) {
        float acc = 0.f;
        for (int d = lane; d < perHead; d += 32) {
            int idx = h * perHead + d;
            float v = ps[idx] + pd[idx];
            v = (v > 0.f) ? v : GAT_SLOPE * v;
            acc += attn[idx] * v;
        }
#pragma unroll
        for (int off = 16; off > 0; off >>= 1)
            acc += __shfl_xor(acc, off, 32);
        if (lane == 0) logits[(long)e * GAT_H + h] = acc;
    }
}

// ---------------- per-node edge softmax + weighted aggregation (one wave32 per node) ----------------
__global__ void node_aggregate(const float* __restrict__ logits,  // [E,H]
                               const float* __restrict__ fs,      // [N,ld]
                               const int* __restrict__ srcIdx,
                               const int* __restrict__ rowptr,
                               const float* __restrict__ resIn,   // residual [N,ld] (may alias out)
                               float* __restrict__ out,           // [N,ld]
                               int N, int ld, int perHead, int doRelu) {
    int wpb = blockDim.x >> 5;
    int node = blockIdx.x * wpb + (threadIdx.x >> 5);
    if (node >= N) return;
    int lane = threadIdx.x & 31;
    int start = rowptr[node], end = rowptr[node + 1];

    // softmax stats: lanes 0..3 each own one head
    float m = -__builtin_inff();
    float s = 0.f;
    if (lane < GAT_H) {
        for (int e = start; e < end; ++e)
            m = fmaxf(m, logits[(long)e * GAT_H + lane]);
        for (int e = start; e < end; ++e)
            s += __expf(logits[(long)e * GAT_H + lane] - m);
    }
    float mh[GAT_H], rsh[GAT_H];
#pragma unroll
    for (int h = 0; h < GAT_H; ++h) {
        mh[h] = __shfl(m, h, 32);
        float sv = __shfl(s, h, 32);
        rsh[h] = (sv > 0.f) ? 1.f / sv : 0.f;
    }

    // feature slots striped across 32 lanes
    const int MAXJ = (GAT_F2 + 31) / 32;   // 6
    float acc[MAXJ];
    int hmap[MAXJ];
#pragma unroll
    for (int j = 0; j < MAXJ; ++j) {
        acc[j] = 0.f;
        int slot = lane + 32 * j;
        hmap[j] = (slot < ld) ? slot / perHead : 0;
    }

    for (int e = start; e < end; ++e) {
        int sn = srcIdx[e];
        const float* fsp = fs + (long)sn * ld;
        float lg = (lane < GAT_H) ? logits[(long)e * GAT_H + lane] : 0.f;
        float wh[GAT_H];
#pragma unroll
        for (int h = 0; h < GAT_H; ++h)
            wh[h] = __expf(__shfl(lg, h, 32) - mh[h]) * rsh[h];
#pragma unroll
        for (int j = 0; j < MAXJ; ++j) {
            int slot = lane + 32 * j;
            if (slot < ld) acc[j] += wh[hmap[j]] * fsp[slot];
        }
    }

#pragma unroll
    for (int j = 0; j < MAXJ; ++j) {
        int slot = lane + 32 * j;
        if (slot < ld) {
            float v = acc[j] + resIn[(long)node * ld + slot];
            if (doRelu) v = fmaxf(v, 0.f);
            out[(long)node * ld + slot] = v;
        }
    }
}

// ---------------- BN column stats (biased var) ----------------
__global__ void bn_stats(const float* __restrict__ x, float* __restrict__ mu,
                         float* __restrict__ var, int N, int ld) {
    int c = blockIdx.x;
    float sum = 0.f, sq = 0.f;
    for (int r = threadIdx.x; r < N; r += blockDim.x) {
        float v = x[(long)r * ld + c];
        sum += v; sq += v * v;
    }
    __shared__ float ss[256], sq2[256];
    ss[threadIdx.x] = sum; sq2[threadIdx.x] = sq;
    __syncthreads();
    for (int off = blockDim.x >> 1; off > 0; off >>= 1) {
        if (threadIdx.x < off) {
            ss[threadIdx.x] += ss[threadIdx.x + off];
            sq2[threadIdx.x] += sq2[threadIdx.x + off];
        }
        __syncthreads();
    }
    if (threadIdx.x == 0) {
        float mean = ss[0] / (float)N;
        mu[c] = mean;
        var[c] = sq2[0] / (float)N - mean * mean;
    }
}

// ---------------- BN apply + ReLU, write fp32 + bf16 ----------------
__global__ void bn_apply_relu_cvt(const float* __restrict__ t,
                                  const float* __restrict__ mu, const float* __restrict__ var,
                                  const float* __restrict__ g,  const float* __restrict__ b,
                                  float* __restrict__ h, __hip_bfloat16* __restrict__ hb,
                                  int N, int ld) {
    long idx = (long)blockIdx.x * blockDim.x + threadIdx.x;
    if (idx >= (long)N * ld) return;
    int c = (int)(idx % ld);
    float v = g[c] * (t[idx] - mu[c]) * rsqrtf(var[c] + GAT_EPS) + b[c];
    v = fmaxf(v, 0.f);
    h[idx] = v;
    hb[idx] = __float2bfloat16(v);
}

// ---------------- mean over heads + log_softmax ----------------
__global__ void head_mean_logsoftmax(const float* __restrict__ t2,  // [N,188]
                                     float* __restrict__ out,       // [N,47]
                                     int N) {
    int n = blockIdx.x * blockDim.x + threadIdx.x;
    if (n >= N) return;
    const float* p = t2 + (long)n * GAT_F2;
    float y[GAT_C];
    float m = -__builtin_inff();
#pragma unroll
    for (int c = 0; c < GAT_C; ++c) {
        float v = 0.25f * (p[c] + p[GAT_C + c] + p[2 * GAT_C + c] + p[3 * GAT_C + c]);
        y[c] = v;
        m = fmaxf(m, v);
    }
    float s = 0.f;
#pragma unroll
    for (int c = 0; c < GAT_C; ++c) s += __expf(y[c] - m);
    float ls = __logf(s);
#pragma unroll
    for (int c = 0; c < GAT_C; ++c) out[(long)n * GAT_C + c] = y[c] - m - ls;
}

// =============================================================================
extern "C" void kernel_launch(void* const* d_in, const int* in_sizes, int n_in,
                              void* d_out, int out_size, void* d_ws, size_t ws_size,
                              hipStream_t stream) {
    const float* x     = (const float*)d_in[0];
    const int*   srcI  = (const int*)d_in[1];
    const int*   dstI  = (const int*)d_in[2];
    const float* Wsrc0 = (const float*)d_in[3];  const float* bsrc0 = (const float*)d_in[4];
    const float* Wdst0 = (const float*)d_in[5];  const float* bdst0 = (const float*)d_in[6];
    const float* attn0 = (const float*)d_in[7];
    const float* Wres0 = (const float*)d_in[8];  const float* bres0 = (const float*)d_in[9];
    const float* Wsrc1 = (const float*)d_in[10]; const float* bsrc1 = (const float*)d_in[11];
    const float* Wdst1 = (const float*)d_in[12]; const float* bdst1 = (const float*)d_in[13];
    const float* attn1 = (const float*)d_in[14];
    const float* Wsrc2 = (const float*)d_in[15]; const float* bsrc2 = (const float*)d_in[16];
    const float* Wdst2 = (const float*)d_in[17]; const float* bdst2 = (const float*)d_in[18];
    const float* attn2 = (const float*)d_in[19];
    const float* Wres2 = (const float*)d_in[20]; const float* bres2 = (const float*)d_in[21];
    const float* g0 = (const float*)d_in[22]; const float* be0 = (const float*)d_in[23];
    const float* g1 = (const float*)d_in[24]; const float* be1 = (const float*)d_in[25];

    const int N = in_sizes[0] / GAT_IN;   // 100000
    const int E = in_sizes[1];            // 1000000

    // ---- workspace bump allocator ----
    char* wp = (char*)d_ws;
    auto alloc = [&](size_t bytes) -> void* {
        void* r = (void*)wp;
        wp += (bytes + 255) & ~(size_t)255;
        return r;
    };
    int*            rowptr = (int*)alloc(((size_t)N + 1) * sizeof(int));
    __hip_bfloat16* xb     = (__hip_bfloat16*)alloc((size_t)N * GAT_KP0 * 2);
    __hip_bfloat16* wtA    = (__hip_bfloat16*)alloc((size_t)GAT_F2P * GAT_KP0 * 2);
    __hip_bfloat16* wtB    = (__hip_bfloat16*)alloc((size_t)GAT_F2P * GAT_KP0 * 2);
    __hip_bfloat16* wtC    = (__hip_bfloat16*)alloc((size_t)GAT_F2P * GAT_KP0 * 2);
    float*          fs     = (float*)alloc((size_t)N * GAT_F2 * 4);
    float*          fd     = (float*)alloc((size_t)N * GAT_F2 * 4);
    float*          res    = (float*)alloc((size_t)N * GAT_F2 * 4);   // residual / agg output (in place)
    float*          h      = (float*)alloc((size_t)N * GAT_F * 4);
    __hip_bfloat16* hb     = (__hip_bfloat16*)alloc((size_t)N * GAT_F * 2);
    float*          logit  = (float*)alloc((size_t)E * GAT_H * 4);
    float*          mu     = (float*)alloc(GAT_F * 4);
    float*          var    = (float*)alloc(GAT_F * 4);

    const int Mtiles = (N + 15) / 16;
    dim3 gemmBlk(128);                          // 4 waves -> 4 M-tile strips per block
    dim3 gemmGrd((Mtiles + 3) / 4);
    const int wtBlk = 256, wtGrd = (GAT_F2P * GAT_KP0 + wtBlk - 1) / wtBlk;
    const int elGrd = (E + 7) / 8;              // 8 waves / block, 1 edge / wave
    const int aggGrd = (N + 7) / 8;             // 8 waves / block, 1 node / wave
    const long nf = (long)N * GAT_F;
    const int bnGrd = (int)((nf + 255) / 256);

    // ---- graph structure ----
    build_rowptr<<<(E + 255) / 256, 256, 0, stream>>>(dstI, rowptr, E, N);

    // ======================= layer 0 =======================
    cvt_pad_bf16<<<(int)(((long)N * GAT_KP0 + 255) / 256), 256, 0, stream>>>(x, xb, N, GAT_IN, GAT_KP0);
    wt_cvt<<<wtGrd, wtBlk, 0, stream>>>(Wsrc0, wtA, GAT_IN, GAT_F, GAT_KP0, GAT_F);
    wt_cvt<<<wtGrd, wtBlk, 0, stream>>>(Wdst0, wtB, GAT_IN, GAT_F, GAT_KP0, GAT_F);
    wt_cvt<<<wtGrd, wtBlk, 0, stream>>>(Wres0, wtC, GAT_IN, GAT_F, GAT_KP0, GAT_F);
    gemm_wmma_bf16_nt<8><<<gemmGrd, gemmBlk, 0, stream>>>(xb, wtA, bsrc0, fs, N, GAT_KP0, GAT_F, GAT_F);
    gemm_wmma_bf16_nt<8><<<gemmGrd, gemmBlk, 0, stream>>>(xb, wtB, bdst0, fd, N, GAT_KP0, GAT_F, GAT_F);
    gemm_wmma_bf16_nt<8><<<gemmGrd, gemmBlk, 0, stream>>>(xb, wtC, bres0, res, N, GAT_KP0, GAT_F, GAT_F);
    edge_logits_wave<<<elGrd, 256, 0, stream>>>(fs, fd, srcI, dstI, attn0, logit, E, GAT_F, GAT_D);
    node_aggregate<<<aggGrd, 256, 0, stream>>>(logit, fs, srcI, rowptr, res, res, N, GAT_F, GAT_D, 1);
    bn_stats<<<GAT_F, 256, 0, stream>>>(res, mu, var, N, GAT_F);
    bn_apply_relu_cvt<<<bnGrd, 256, 0, stream>>>(res, mu, var, g0, be0, h, hb, N, GAT_F);

    // ======================= layer 1 (identity residual = h) =======================
    wt_cvt<<<wtGrd, wtBlk, 0, stream>>>(Wsrc1, wtA, GAT_F, GAT_F, GAT_KP0, GAT_F);
    wt_cvt<<<wtGrd, wtBlk, 0, stream>>>(Wdst1, wtB, GAT_F, GAT_F, GAT_KP0, GAT_F);
    gemm_wmma_bf16_nt<8><<<gemmGrd, gemmBlk, 0, stream>>>(hb, wtA, bsrc1, fs, N, GAT_KP0, GAT_F, GAT_F);
    gemm_wmma_bf16_nt<8><<<gemmGrd, gemmBlk, 0, stream>>>(hb, wtB, bdst1, fd, N, GAT_KP0, GAT_F, GAT_F);
    edge_logits_wave<<<elGrd, 256, 0, stream>>>(fs, fd, srcI, dstI, attn1, logit, E, GAT_F, GAT_D);
    node_aggregate<<<aggGrd, 256, 0, stream>>>(logit, fs, srcI, rowptr, h, res, N, GAT_F, GAT_D, 1);
    bn_stats<<<GAT_F, 256, 0, stream>>>(res, mu, var, N, GAT_F);
    bn_apply_relu_cvt<<<bnGrd, 256, 0, stream>>>(res, mu, var, g1, be1, h, hb, N, GAT_F);

    // ======================= layer 2 (Nc = 188) =======================
    wt_cvt<<<wtGrd, wtBlk, 0, stream>>>(Wsrc2, wtA, GAT_F, GAT_F2, GAT_KP0, GAT_F2P);
    wt_cvt<<<wtGrd, wtBlk, 0, stream>>>(Wdst2, wtB, GAT_F, GAT_F2, GAT_KP0, GAT_F2P);
    wt_cvt<<<wtGrd, wtBlk, 0, stream>>>(Wres2, wtC, GAT_F, GAT_F2, GAT_KP0, GAT_F2P);
    gemm_wmma_bf16_nt<12><<<gemmGrd, gemmBlk, 0, stream>>>(hb, wtA, bsrc2, fs, N, GAT_KP0, GAT_F2, GAT_F2);
    gemm_wmma_bf16_nt<12><<<gemmGrd, gemmBlk, 0, stream>>>(hb, wtB, bdst2, fd, N, GAT_KP0, GAT_F2, GAT_F2);
    gemm_wmma_bf16_nt<12><<<gemmGrd, gemmBlk, 0, stream>>>(hb, wtC, bres2, res, N, GAT_KP0, GAT_F2, GAT_F2);
    edge_logits_wave<<<elGrd, 256, 0, stream>>>(fs, fd, srcI, dstI, attn2, logit, E, GAT_F2, GAT_C);
    node_aggregate<<<aggGrd, 256, 0, stream>>>(logit, fs, srcI, rowptr, res, res, N, GAT_F2, GAT_C, 0);

    head_mean_logsoftmax<<<(N + 255) / 256, 256, 0, stream>>>(res, (float*)d_out, N);
}
